// CustomCrossAttentionExt2_86569360818814
// MI455X (gfx1250) — compile-verified, matrix operand order checked by reading, not run on hardware
//
#include <hip/hip_runtime.h>
#include <hip/hip_bf16.h>
#include <math.h>

// ---------------------------------------------------------------------------
// Problem constants
// ---------------------------------------------------------------------------
#define BB   4
#define NN   4096
#define JJ   77
#define HH   16
#define DHH  64
#define DQQ  1024
#define DCC  768
#define DHID 384           // DC/2
#define INNERD 1024        // H*DH
#define MROWS (BB*NN)      // 16384
#define JPAD 80            // sim col padding (5 x 16)
#define KPAD 96            // attn K padding  (3 x 32)

typedef __attribute__((ext_vector_type(16))) _Float16 v16h;
typedef __attribute__((ext_vector_type(8)))  float    v8f;
typedef __attribute__((ext_vector_type(2)))  _Float16 h2_t;

union Frag16 { v16h v; h2_t h2[8]; _Float16 h[16]; };

// load one 16x32 f16 WMMA operand fragment from LDS (row-major, "base" points
// at this lane's row at the k-step origin; ISA 16-bit A/B layout)
__device__ __forceinline__ void load_frag(const _Float16* base, Frag16& f, int lhi)
{
    #pragma unroll
    for (int r = 0; r < 8; ++r) {
        int kk = (r < 4) ? (lhi * 8 + 2 * r) : (16 + lhi * 8 + 2 * (r - 4));
        f.h2[r] = *(const h2_t*)(base + kk);
    }
}

// ---------------------------------------------------------------------------
// Kernel 1: token-type embedding MLP -> kin = contextembs + typeemb
// one block per (b,j) row (tiny: 0.4 GFLOP total)
// ---------------------------------------------------------------------------
__global__ __launch_bounds__(256)
void prep_kernel(const float* __restrict__ ce, const int* __restrict__ ct,
                 const float* __restrict__ tte_emb,
                 const float* __restrict__ m1w, const float* __restrict__ m1b,
                 const float* __restrict__ m2w, const float* __restrict__ m2b,
                 const float* __restrict__ gA, const float* __restrict__ gB,
                 float* __restrict__ kin)
{
    const int bj = blockIdx.x;            // 0..307
    const int tid = threadIdx.x;
    __shared__ float tte_s[DCC];
    __shared__ float cein_s[DCC];
    __shared__ float hid_s[DHID];

    int t = ct[bj];
    int tcl = t < 0 ? 0 : (t > 4 ? 4 : t);

    for (int c = tid; c < DCC; c += 256) {
        float tv = tte_emb[(size_t)tcl * DCC + c];
        float ev = ce[(size_t)bj * DCC + c];
        tte_s[c]  = tv;
        cein_s[c] = ev + tv;
    }
    __syncthreads();

    for (int col = tid; col < DHID; col += 256) {
        float acc = m1b[col];
        for (int c = 0; c < DCC; ++c)
            acc = fmaf(cein_s[c], m1w[(size_t)c * DHID + col], acc);
        hid_s[col] = 0.5f * acc * (1.0f + erff(acc * 0.70710678118654752f));
    }
    __syncthreads();

    for (int col = tid; col < DCC; col += 256) {
        float acc = m2b[col];
        for (int c = 0; c < DHID; ++c)
            acc = fmaf(hid_s[c], m2w[(size_t)c * DCC + col], acc);
        float typeemb = tte_s[col] * gA[col] + acc * gB[col];
        kin[(size_t)bj * DCC + col] = ce[(size_t)bj * DCC + col] + typeemb;
    }
}

// ---------------------------------------------------------------------------
// WMMA GEMM: C[M,1024] = A[M,K] * B[K,1024]  (fp32 in, f16 mul, f32 acc)
// Block tile 128x128, BK=32, 8 waves (4x2), each wave 32x64 = 2x4 WMMA tiles.
// MODE 0: store q permuted [B*H, N, DH]
// MODE 1: plain store + bias (output projection)
// MODE 2: store K/V permuted [B*H, J, DH]  (rows = b*J+j, M=308)
// ---------------------------------------------------------------------------
#define GBM 128
#define GBN 128
#define GBK 32
#define LDSS 40   // padded LDS stride in halfs

template<int MODE>
__global__ __launch_bounds__(256)
void wmma_gemm(const float* __restrict__ A, const float* __restrict__ Bm,
               const float* __restrict__ bias, float* __restrict__ C,
               int M, int K)
{
    __shared__ _Float16 Ash[GBM * LDSS];
    __shared__ _Float16 Bsh[GBN * LDSS];

    const int tid  = threadIdx.x;
    const int wid  = tid >> 5;
    const int lane = tid & 31;
    const int wm   = wid >> 1;     // 0..3 -> 32-row strip
    const int wn   = wid & 1;      // 0..1 -> 64-col strip
    const int lm   = lane & 15;
    const int lhi  = lane >> 4;

    const int blockM = blockIdx.y * GBM;
    const int blockN = blockIdx.x * GBN;

    v8f acc[2][4];
    #pragma unroll
    for (int t = 0; t < 2; ++t)
        #pragma unroll
        for (int u = 0; u < 4; ++u)
            acc[t][u] = (v8f){0.f,0.f,0.f,0.f,0.f,0.f,0.f,0.f};

    int arow = blockM + (tid >> 1);       // global A row
    if (arow >= M) arow = M - 1;          // clamp (stores are guarded)
    const int acb  = (tid & 1) * 16;
    const int bkr  = tid >> 3;            // 0..31
    const int bcb  = (tid & 7) * 16;

    for (int k0 = 0; k0 < K; k0 += GBK) {
        { // stage A tile (fp32 -> f16)
            const float* src = A + (size_t)arow * K + k0 + acb;
            _Float16* dst = &Ash[(tid >> 1) * LDSS + acb];
            #pragma unroll
            for (int i = 0; i < 16; ++i) dst[i] = (_Float16)src[i];
        }
        { // stage B tile transposed: Bsh[n][k]
            const float* src = Bm + (size_t)(k0 + bkr) * 1024 + blockN + bcb;
            #pragma unroll
            for (int i = 0; i < 16; ++i)
                Bsh[(bcb + i) * LDSS + bkr] = (_Float16)src[i];
        }
        if (k0 + GBK < K) {   // global_prefetch of next k-tile
            __builtin_prefetch(A + (size_t)arow * K + k0 + GBK + acb, 0, 1);
            __builtin_prefetch(Bm + (size_t)(k0 + GBK + bkr) * 1024 + blockN + bcb, 0, 1);
        }
        __syncthreads();

        Frag16 af[2], bf[4];
        #pragma unroll
        for (int t = 0; t < 2; ++t)
            load_frag(&Ash[(wm * 32 + t * 16 + lm) * LDSS], af[t], lhi);
        #pragma unroll
        for (int u = 0; u < 4; ++u)
            load_frag(&Bsh[(wn * 64 + u * 16 + lm) * LDSS], bf[u], lhi);

        #pragma unroll
        for (int t = 0; t < 2; ++t)
            #pragma unroll
            for (int u = 0; u < 4; ++u)
                acc[t][u] = __builtin_amdgcn_wmma_f32_16x16x32_f16(
                    false, af[t].v, false, bf[u].v,
                    (short)0, acc[t][u], false, false);

        __syncthreads();
    }

    // C/D layout: lanes 0-15 -> M=r, lanes 16-31 -> M=8+r, N=lane&15
    #pragma unroll
    for (int t = 0; t < 2; ++t) {
        #pragma unroll
        for (int u = 0; u < 4; ++u) {
            #pragma unroll
            for (int r = 0; r < 8; ++r) {
                int rg = blockM + wm * 32 + t * 16 + (lhi ? 8 + r : r);
                int cg = blockN + wn * 64 + u * 16 + lm;
                float val = acc[t][u][r];
                if (MODE == 1) {
                    C[(size_t)rg * 1024 + cg] = val + bias[cg];
                } else if (MODE == 0) {
                    int b = rg >> 12, n = rg & 4095;
                    int h = cg >> 6,  d = cg & 63;
                    C[(((size_t)(b * HH + h)) * NN + n) * DHH + d] = val;
                } else { // MODE 2: rows are (b, j)
                    if (rg < M) {
                        int b = rg / JJ, j = rg - b * JJ;
                        int h = cg >> 6, d = cg & 63;
                        C[(((size_t)(b * HH + h)) * JJ + j) * DHH + d] = val;
                    }
                }
            }
        }
    }
}

// ---------------------------------------------------------------------------
// Kernel 3: WMMA sim tiles + masked sum/sumsq (sim never hits HBM)
// grid (N/128, B*H), 256 threads (8 waves x 16 rows), 5 j-tiles, K=64
// ---------------------------------------------------------------------------
__global__ __launch_bounds__(256)
void sim_stats_kernel(const float* __restrict__ q_bh, const float* __restrict__ k_bh,
                      const int* __restrict__ ct, float* __restrict__ stats)
{
    __shared__ _Float16 qsh[128 * 72];     // [row][k], stride 72
    __shared__ _Float16 kshT[JPAD * 72];   // [j][k], rows 77..79 zero
    __shared__ int   sel2[JPAD];
    __shared__ float rs[256], rq[256];

    const int tid  = threadIdx.x;
    const int wid  = tid >> 5;
    const int lane = tid & 31;
    const int lm   = lane & 15;
    const int lhi  = lane >> 4;
    const int bh   = blockIdx.y;
    const int b    = bh >> 4;
    const int rowbase = blockIdx.x * 128;

    { // q tile: 128x64 fp32 -> f16
        int r = tid >> 1, cb = (tid & 1) * 32;
        const float* src = q_bh + ((size_t)bh * NN + rowbase + r) * DHH + cb;
        #pragma unroll
        for (int i = 0; i < 32; ++i) qsh[r * 72 + cb + i] = (_Float16)src[i];
    }
    for (int idx = tid; idx < JPAD * DHH; idx += 256) {
        int j = idx >> 6, k = idx & 63;
        float v = (j < JJ) ? k_bh[((size_t)bh * JJ + j) * DHH + k] : 0.f;
        kshT[j * 72 + k] = (_Float16)v;
    }
    for (int j = tid; j < JPAD; j += 256)
        sel2[j] = (j < JJ && ct[b * JJ + j] >= 2) ? 1 : 0;
    __syncthreads();

    v8f acc[5];
    #pragma unroll
    for (int t = 0; t < 5; ++t) acc[t] = (v8f){0.f,0.f,0.f,0.f,0.f,0.f,0.f,0.f};

    #pragma unroll
    for (int s = 0; s < 2; ++s) {          // K = 64 -> 2 k-steps
        Frag16 af;
        load_frag(&qsh[(wid * 16 + lm) * 72 + s * 32], af, lhi);
        #pragma unroll
        for (int t = 0; t < 5; ++t) {
            Frag16 bf;
            load_frag(&kshT[(t * 16 + lm) * 72 + s * 32], bf, lhi);
            acc[t] = __builtin_amdgcn_wmma_f32_16x16x32_f16(
                false, af.v, false, bf.v, (short)0, acc[t], false, false);
        }
    }

    // lane (t,lm) owns column j = t*16+lm for 8 rows -> accumulate stats
    float s1 = 0.f, s2 = 0.f;
    #pragma unroll
    for (int t = 0; t < 5; ++t) {
        int j = t * 16 + lm;
        if (j < JJ && sel2[j]) {
            #pragma unroll
            for (int r = 0; r < 8; ++r) {
                float v = acc[t][r];
                s1 += v;
                s2 = fmaf(v, v, s2);
            }
        }
    }
    rs[tid] = s1; rq[tid] = s2;
    __syncthreads();
    for (int off = 128; off > 0; off >>= 1) {
        if (tid < off) { rs[tid] += rs[tid + off]; rq[tid] += rq[tid + off]; }
        __syncthreads();
    }
    if (tid == 0) {
        atomicAdd(&stats[0], rs[0]);
        atomicAdd(&stats[1], rq[0]);
    }
}

// ---------------------------------------------------------------------------
// Kernel 4: finalize std (Bessel-corrected, matching torch .std())
// ---------------------------------------------------------------------------
__global__ void finalize_kernel(const int* __restrict__ ct, float* __restrict__ stats)
{
    int cnt = 0;
    for (int i = 0; i < BB * JJ; ++i) cnt += (ct[i] >= 2) ? 1 : 0;
    float nsel = (float)cnt * (float)HH * (float)NN;
    float mean = (nsel > 0.f) ? stats[0] / nsel : 0.f;
    float var  = (nsel > 1.f) ? (stats[1] - nsel * mean * mean) / (nsel - 1.f) : 0.f;
    stats[2] = sqrtf(var > 0.f ? var : 0.f);
}

// ---------------------------------------------------------------------------
// Kernel 5: fused attention block, 64 rows / block, 128 threads (4 waves):
//   WMMA sim -> LDS -> masked softmax (VALU, unnormalized f16 probs) ->
//   WMMA attn@V with 1/denom folded into the store.
// LDS kept < 64 KB via union aliasing of the q tile with the attn tile.
// ---------------------------------------------------------------------------
__global__ __launch_bounds__(128)
void attn_fused_kernel(const float* __restrict__ q_bh, const float* __restrict__ k_bh,
                       const float* __restrict__ v_bh, const int* __restrict__ ct,
                       const float* __restrict__ camask, const float* __restrict__ strength,
                       const float* __restrict__ stats, float* __restrict__ out_attn)
{
    __shared__ union {
        _Float16 qsh[64 * 72];         // stage 1: q tile [row][k]
        _Float16 attn_sh[64 * 104];    // stage 2: probs [row][j], j<96, zero-pad
    } u;
    __shared__ _Float16 kshT[JPAD * 72];   // [j][k], rows 77..79 zero
    __shared__ _Float16 vshT[64 * 104];    // [d][k], k 77..95 zero
    __shared__ float    simsh[64 * 81];    // sim tile fp32, stride 81
    __shared__ float    invs[64];
    __shared__ int      sel2[JPAD], selg[JPAD];

    const int tid  = threadIdx.x;
    const int wid  = tid >> 5;     // 0..3 -> 16-row strip
    const int lane = tid & 31;
    const int lm   = lane & 15;
    const int lhi  = lane >> 4;
    const int bh   = blockIdx.y;
    const int b    = bh >> 4;
    const int h    = bh & 15;
    const int rowbase = blockIdx.x * 64;

    { // q tile 64x64 fp32 -> f16
        int r = tid >> 1, cb = (tid & 1) * 32;
        const float* src = q_bh + ((size_t)bh * NN + rowbase + r) * DHH + cb;
        #pragma unroll
        for (int i = 0; i < 32; ++i) u.qsh[r * 72 + cb + i] = (_Float16)src[i];
    }
    for (int idx = tid; idx < JPAD * DHH; idx += 128) {
        int j = idx >> 6, k = idx & 63;
        float v = (j < JJ) ? k_bh[((size_t)bh * JJ + j) * DHH + k] : 0.f;
        kshT[j * 72 + k] = (_Float16)v;
    }
    for (int idx = tid; idx < DHH * KPAD; idx += 128) {
        int d = idx / KPAD, k = idx - d * KPAD;
        float v = (k < JJ) ? v_bh[((size_t)bh * JJ + k) * DHH + d] : 0.f;
        vshT[d * 104 + k] = (_Float16)v;
    }
    for (int j = tid; j < JPAD; j += 128) {
        int t = (j < JJ) ? ct[b * JJ + j] : -1;
        sel2[j] = (t >= 2) ? 1 : 0;
        selg[j] = (t >= 0 && t < 2) ? 1 : 0;
    }
    __syncthreads();

    // ---- stage 1: sim = q @ k^T via WMMA, into LDS
    {
        v8f acc[5];
        #pragma unroll
        for (int t = 0; t < 5; ++t) acc[t] = (v8f){0.f,0.f,0.f,0.f,0.f,0.f,0.f,0.f};
        #pragma unroll
        for (int s = 0; s < 2; ++s) {
            Frag16 af;
            load_frag(&u.qsh[(wid * 16 + lm) * 72 + s * 32], af, lhi);
            #pragma unroll
            for (int t = 0; t < 5; ++t) {
                Frag16 bf;
                load_frag(&kshT[(t * 16 + lm) * 72 + s * 32], bf, lhi);
                acc[t] = __builtin_amdgcn_wmma_f32_16x16x32_f16(
                    false, af.v, false, bf.v, (short)0, acc[t], false, false);
            }
        }
        #pragma unroll
        for (int t = 0; t < 5; ++t) {
            int j = t * 16 + lm;
            #pragma unroll
            for (int r = 0; r < 8; ++r) {
                int rl = wid * 16 + (lhi ? 8 + r : r);
                simsh[rl * 81 + j] = acc[t][r];
            }
        }
    }
    __syncthreads();

    // ---- stage 2: masked softmax per row (writes f16 probs over the q tile)
    if (tid < 64) {
        const int rl = tid;
        const int rowg = rowbase + rl;
        const float simstd = stats[2];
        const float str    = strength[0];
        const float scale  = 0.125f;                  // 1/sqrt(DH)
        const float* camrow = camask + ((size_t)b * NN + rowg) * JJ;

        float wfmax = -__FLT_MAX__;
        for (int j = 0; j < JJ; ++j) {
            float wf = sel2[j] ? camrow[j] * simstd * str : 0.f;
            wfmax = fmaxf(wfmax, wf);
        }
        const float thr = fmaxf(wfmax, 0.001f) - 0.0001f;

        float m = -__FLT_MAX__;
        for (int j = 0; j < JJ; ++j) {
            float wf = sel2[j] ? camrow[j] * simstd * str : 0.f;
            bool lmv = (wf >= thr) || (selg[j] != 0);
            float sv = lmv ? simsh[rl * 81 + j] * scale : -__FLT_MAX__;
            simsh[rl * 81 + j] = sv;                  // stash masked+scaled score
            m = fmaxf(m, sv);
        }
        float denom = 0.f;
        for (int j = 0; j < JJ; ++j) {
            float p = __expf(simsh[rl * 81 + j] - m);
            u.attn_sh[rl * 104 + j] = (_Float16)p;    // unnormalized
            denom += p;
        }
        #pragma unroll
        for (int j = JJ; j < KPAD; ++j) u.attn_sh[rl * 104 + j] = (_Float16)0.f;
        invs[rl] = 1.f / denom;
    }
    __syncthreads();

    // ---- stage 3: out = attn @ V via WMMA (K = 96 -> 3 k-steps, 4 col tiles)
    {
        v8f acc[4];
        #pragma unroll
        for (int t = 0; t < 4; ++t) acc[t] = (v8f){0.f,0.f,0.f,0.f,0.f,0.f,0.f,0.f};
        #pragma unroll
        for (int s = 0; s < 3; ++s) {
            Frag16 af;
            load_frag(&u.attn_sh[(wid * 16 + lm) * 104 + s * 32], af, lhi);
            #pragma unroll
            for (int t = 0; t < 4; ++t) {
                Frag16 bf;
                load_frag(&vshT[(t * 16 + lm) * 104 + s * 32], bf, lhi);
                acc[t] = __builtin_amdgcn_wmma_f32_16x16x32_f16(
                    false, af.v, false, bf.v, (short)0, acc[t], false, false);
            }
        }
        #pragma unroll
        for (int t = 0; t < 4; ++t) {
            int d = t * 16 + lm;
            #pragma unroll
            for (int r = 0; r < 8; ++r) {
                int rl = wid * 16 + (lhi ? 8 + r : r);
                int rowg = rowbase + rl;
                out_attn[((size_t)b * NN + rowg) * INNERD + h * DHH + d] =
                    acc[t][r] * invs[rl];
            }
        }
    }
}

// ---------------------------------------------------------------------------
// launch
// ---------------------------------------------------------------------------
extern "C" void kernel_launch(void* const* d_in, const int* in_sizes, int n_in,
                              void* d_out, int out_size, void* d_ws, size_t ws_size,
                              hipStream_t stream)
{
    const float* x        = (const float*)d_in[0];
    const float* ce       = (const float*)d_in[1];
    const int*   ct       = (const int*)  d_in[2];
    const float* camask   = (const float*)d_in[3];
    const float* strength = (const float*)d_in[4];
    const float* Wq       = (const float*)d_in[5];
    const float* Wk       = (const float*)d_in[6];
    const float* Wv       = (const float*)d_in[7];
    const float* Wout     = (const float*)d_in[8];
    const float* bout     = (const float*)d_in[9];
    const float* tte      = (const float*)d_in[10];
    const float* m1w      = (const float*)d_in[11];
    const float* m1b      = (const float*)d_in[12];
    const float* m2w      = (const float*)d_in[13];
    const float* m2b      = (const float*)d_in[14];
    const float* gA       = (const float*)d_in[15];
    const float* gB       = (const float*)d_in[16];
    float* out = (float*)d_out;

    // workspace partition (floats)
    float* ws    = (float*)d_ws;
    float* kin   = ws;                                  // 308*768
    float* k_bh  = kin  + (size_t)BB*JJ*DCC;            // 64*77*64
    float* v_bh  = k_bh + (size_t)BB*HH*JJ*DHH;
    float* q_bh  = v_bh + (size_t)BB*HH*JJ*DHH;         // 64*4096*64
    float* aout  = q_bh + (size_t)BB*HH*NN*DHH;         // 16384*1024
    float* stats = aout + (size_t)MROWS*INNERD;         // [sum, sumsq, std]

    hipMemsetAsync(stats, 0, 4 * sizeof(float), stream);

    prep_kernel<<<BB*JJ, 256, 0, stream>>>(ce, ct, tte, m1w, m1b, m2w, m2b, gA, gB, kin);

    // K = kin @ Wk, V = ce @ Wv  (WMMA, M=308, K=768, permuted -> [B*H, J, DH])
    wmma_gemm<2><<<dim3(INNERD/GBN, (BB*JJ + GBM - 1)/GBM), 256, 0, stream>>>(
        kin, Wk, nullptr, k_bh, BB*JJ, DCC);
    wmma_gemm<2><<<dim3(INNERD/GBN, (BB*JJ + GBM - 1)/GBM), 256, 0, stream>>>(
        ce, Wv, nullptr, v_bh, BB*JJ, DCC);

    // q = x @ Wq  (WMMA, permuted -> [B*H, N, DH])
    wmma_gemm<0><<<dim3(INNERD/GBN, MROWS/GBM), 256, 0, stream>>>(
        x, Wq, nullptr, q_bh, MROWS, DQQ);

    // masked mean/std of sim (WMMA sim tiles, never stored)
    sim_stats_kernel<<<dim3(NN/128, BB*HH), 256, 0, stream>>>(q_bh, k_bh, ct, stats);
    finalize_kernel<<<1, 1, 0, stream>>>(ct, stats);

    // fused WMMA sim -> softmax -> WMMA attn@V
    attn_fused_kernel<<<dim3(NN/64, BB*HH), 128, 0, stream>>>(
        q_bh, k_bh, v_bh, ct, camask, strength, stats, aout);

    // out = attn_out @ Wout + bout  (WMMA)
    wmma_gemm<1><<<dim3(DQQ/GBN, MROWS/GBM), 256, 0, stream>>>(
        aout, Wout, bout, out, MROWS, DQQ);
}
